// Model_58978490909154
// MI455X (gfx1250) — compile-verified, matrix operand order checked by reading, not run on hardware
//
#include <hip/hip_runtime.h>
#include <hip/hip_bf16.h>
#include <math.h>

typedef __attribute__((ext_vector_type(16))) _Float16 v16h;
typedef __attribute__((ext_vector_type(8)))  float    v8f;

// ---------------- workspace layout (float offsets) ----------------
#define ACC_CNT   0      // [3]
#define ACC_REG   3      // [3]
#define ACC_CLS   6      // [3]
#define ACC_SP    9      // [3] sum softplus(p_obj)
#define ACC_CORR  12     // [3] sum p_obj * pos
#define POS0      16                 // 16*3*6400 = 307200
#define POS1      307216             // 16*3*1600 =  76800
#define POS2      384016             // 16*3*400  =  19200
#define ZERO_END  403216
#define POBJ0     403216
#define POBJ1     710416
#define POBJ2     787216
#define META0     806416             // 7680 * 12 floats per level
#define META1     898576
#define META2     990736
#define PVOFF     1082896            // 7680 * 96 floats (reused per level)

__constant__ float c_anchors[3][3][2] = {
  {{1.25f,1.625f},{2.0f,3.75f},{4.125f,2.875f}},
  {{1.875f,3.8125f},{3.875f,2.8125f},{3.6875f,7.4375f}},
  {{3.625f,2.8125f},{4.875f,6.1875f},{11.65625f,10.1875f}}};

__device__ __forceinline__ float softplusf(float x){
  return fmaxf(x,0.f) + log1pf(expf(-fabsf(x)));
}
__device__ __forceinline__ float sigmf(float x){ return 1.f/(1.f+expf(-x)); }

__device__ float ciou_f(float x1,float y1,float w1,float h1,
                        float x2,float y2,float w2,float h2){
  const float eps = 1e-7f;
  float b1x1=x1-w1*0.5f, b1x2=x1+w1*0.5f, b1y1=y1-h1*0.5f, b1y2=y1+h1*0.5f;
  float b2x1=x2-w2*0.5f, b2x2=x2+w2*0.5f, b2y1=y2-h2*0.5f, b2y2=y2+h2*0.5f;
  float iw = fmaxf(fminf(b1x2,b2x2)-fmaxf(b1x1,b2x1), 0.f);
  float ih = fmaxf(fminf(b1y2,b2y2)-fmaxf(b1y1,b2y1), 0.f);
  float inter = iw*ih;
  float uni = w1*h1 + w2*h2 - inter + eps;
  float iou = inter/uni;
  float cw = fmaxf(b1x2,b2x2)-fminf(b1x1,b2x1);
  float ch = fmaxf(b1y2,b2y2)-fminf(b1y1,b2y1);
  float c2 = cw*cw + ch*ch + eps;
  float dxs = b2x1+b2x2-b1x1-b1x2, dys = b2y1+b2y2-b1y1-b1y2;
  float rho2 = (dxs*dxs + dys*dys)*0.25f;
  float dv = atanf(w2/(h2+eps)) - atanf(w1/(h1+eps));
  float v  = (4.0f/(3.14159265358979f*3.14159265358979f))*dv*dv;
  float alpha = v/(v - iou + 1.0f + eps);
  return iou - (rho2/c2 + v*alpha);
}

__global__ void init_kernel(float* ws){
  int i = blockIdx.x*blockDim.x + threadIdx.x;
  if (i < ZERO_END) ws[i] = 0.f;
}

// ---- streaming pass: p_obj for 3 anchors over full map + sum softplus ----
// One wave handles 16 consecutive spatial positions; WMMA M-tile has 3 live
// weight rows (obj channel of each anchor), rest zero.  Memory-bound.
__global__ void obj_map_kernel(const float* __restrict__ f, const float* __restrict__ w,
                               const float* __restrict__ bias, float* __restrict__ pobj,
                               float* __restrict__ acc_sp, int C, int HW){
  int wid  = (blockIdx.x*blockDim.x + threadIdx.x) >> 5;
  int lane = threadIdx.x & 31;
  int tilesPerB = HW >> 4;
  int b   = wid / tilesPerB;
  int hwb = (wid % tilesPerB) << 4;
  int n   = lane & 15;
  int kbl = (lane >> 4) << 4;          // 0 or 16 (K half per lane group)

  union CU { v8f v; float s[8]; } cu;
  #pragma unroll
  for (int i=0;i<8;i++) cu.s[i]=0.f;

  int ksteps = C >> 5;
  for (int ks=0; ks<ksteps; ks++){
    int kb = kbl + (ks<<5);
    union HU { v16h v; _Float16 h[16]; } A, Bv;
    if (n < 3){
      const float* wr = w + (n*85+4)*C + kb;
      #pragma unroll
      for (int t=0;t<16;t++) A.h[t] = (_Float16)wr[t];
    } else {
      #pragma unroll
      for (int t=0;t<16;t++) A.h[t] = (_Float16)0.0f;
    }
    const float* fp = f + (size_t)(b*C + kb)*HW + hwb + n;
    __builtin_prefetch(fp + (size_t)32*HW, 0, 1);      // next K-step
    #pragma unroll
    for (int t=0;t<16;t++) Bv.h[t] = (_Float16)fp[(size_t)t*HW];
    cu.v = __builtin_amdgcn_wmma_f32_16x16x32_f16(false, A.v, false, Bv.v,
                                                  (short)0, cu.v, false, false);
  }
  float sp = 0.f;
  if (lane < 16){
    #pragma unroll
    for (int a_=0;a_<3;a_++){
      float x = cu.s[a_] + bias[a_*85+4];
      pobj[(size_t)(b*3 + a_)*HW + hwb + n] = x;
      sp += softplusf(x);
    }
  }
  for (int off=16; off>0; off>>=1) sp += __shfl_xor(sp, off, 32);
  if (lane == 0) atomicAdd(acc_sp, sp);
}

// ---- target assignment: one thread per (level, candidate) ----
__global__ void targets_kernel(const float* __restrict__ gt_boxes,
                               const int* __restrict__ gt_cls,
                               const int* __restrict__ img_sz,
                               float* __restrict__ ws){
  int idx = blockIdx.x*blockDim.x + threadIdx.x;
  if (idx >= 3*7680) return;
  const int metaoff[3] = {META0, META1, META2};
  int lvl = idx / 7680, cand = idx % 7680;
  int o = cand / 1536, rem = cand % 1536;
  int a = rem / 512,   j   = rem % 512;
  int b = j / 32,      t   = j % 32;
  int Wd = (lvl==0)?80:((lvl==1)?40:20);
  float img = (float)img_sz[0];
  const float* gb = gt_boxes + ((b*32 + t)<<2);
  float cx=(gb[0]+gb[2])*0.5f/img, cy=(gb[1]+gb[3])*0.5f/img;
  float gw=(gb[2]-gb[0])/img*Wd,  gh=(gb[3]-gb[1])/img*Wd;
  float gx=cx*Wd, gy=cy*Wd;
  float aw=c_anchors[lvl][a][0], ah=c_anchors[lvl][a][1];
  float rw=gw/aw, rh=gh/ah;
  float mr=fmaxf(fmaxf(rw,1.f/rw), fmaxf(rh,1.f/rh));
  float m0=(mr < 4.0f)?1.f:0.f;
  float fx=gx-floorf(gx), fy=gy-floorf(gy);
  bool jkx=(fx<0.5f)&&(gx>1.0f), jky=(fy<0.5f)&&(gy>1.0f);
  float ix=(float)Wd-gx, iy=(float)Wd-gy;
  float fix=ix-floorf(ix), fiy=iy-floorf(iy);
  bool lmx=(fix<0.5f)&&(ix>1.0f), lmy=(fiy<0.5f)&&(iy>1.0f);
  float rowm = (o==0)?1.f:(o==1)?(jkx?1.f:0.f):(o==2)?(jky?1.f:0.f)
              :(o==3)?(lmx?1.f:0.f):(lmy?1.f:0.f);
  float mask = rowm * m0;
  float ox=(o==1)?0.5f:((o==3)?-0.5f:0.f);
  float oy=(o==2)?0.5f:((o==4)?-0.5f:0.f);
  float gijx=truncf(gx-ox), gijy=truncf(gy-oy);
  int gi = min(max((int)gijx,0), Wd-1);   // ref clips x by H-1 (H==W here)
  int gj = min(max((int)gijy,0), Wd-1);
  float* mrow = ws + metaoff[lvl] + (size_t)cand*12;
  mrow[0]=mask; mrow[1]=(float)b; mrow[2]=(float)a;
  mrow[3]=(float)gj; mrow[4]=(float)gi;
  mrow[5]=gx-gijx; mrow[6]=gy-gijy; mrow[7]=gw; mrow[8]=gh;
  mrow[9]=(float)gt_cls[b*32+t];
  atomicAdd(ws + ACC_CNT + lvl, mask);
}

// ---- WMMA scatter-GEMM: 96 output rows (anchor slice) per 16 candidates ----
__global__ void pos_gemm_kernel(const float* __restrict__ f, const float* __restrict__ w,
                                const float* __restrict__ bias, const float* __restrict__ meta,
                                float* __restrict__ pv, int C, int HW, int Wd){
  int wid  = (blockIdx.x*blockDim.x + threadIdx.x) >> 5;
  int lane = threadIdx.x & 31;
  int ct = wid / 6, mt = wid % 6;
  int cbase = ct << 4;
  int a = (cbase % 1536) / 512;        // whole tile shares this anchor
  int mbase = a*85 + mt*16;
  int n   = lane & 15;
  int kbl = (lane >> 4) << 4;
  int cand = cbase + n;
  const float* mrow = meta + (size_t)cand*12;
  int bi=(int)mrow[1], gj=(int)mrow[3], gi=(int)mrow[4];
  size_t fbase = (size_t)bi*C*HW + (size_t)gj*Wd + gi;
  int mg_lane = mbase + n;

  union CU { v8f v; float s[8]; } cu;
  #pragma unroll
  for (int i=0;i<8;i++) cu.s[i]=0.f;

  int ksteps = C >> 5;
  for (int ks=0; ks<ksteps; ks++){
    int kb = kbl + (ks<<5);
    union HU { v16h v; _Float16 h[16]; } A, Bv;
    if (mg_lane < 255){
      const float* wr = w + (size_t)mg_lane*C + kb;
      #pragma unroll
      for (int t=0;t<16;t++) A.h[t]=(_Float16)wr[t];
    } else {
      #pragma unroll
      for (int t=0;t<16;t++) A.h[t]=(_Float16)0.0f;
    }
    const float* fp = f + fbase + (size_t)kb*HW;
    #pragma unroll
    for (int t=0;t<16;t++) Bv.h[t]=(_Float16)fp[(size_t)t*HW];
    cu.v = __builtin_amdgcn_wmma_f32_16x16x32_f16(false, A.v, false, Bv.v,
                                                  (short)0, cu.v, false, false);
  }
  int hi = (lane >= 16) ? 8 : 0;
  #pragma unroll
  for (int r=0;r<8;r++){
    int mg = mbase + r + hi;
    float bv = (mg < 255) ? bias[mg] : 0.f;
    pv[(size_t)cand*96 + mt*16 + r + hi] = cu.s[r] + bv;
  }
}

// ---- per-candidate losses: CIoU reg, cls BCE, scatter-max into pos ----
__global__ void loss_pos_kernel(const float* __restrict__ meta, const float* __restrict__ pv,
                                float* __restrict__ pos, float* __restrict__ ws,
                                int lvl, int Wd){
  int cand = blockIdx.x*blockDim.x + threadIdx.x;
  if (cand >= 7680) return;
  const float* mrow = meta + (size_t)cand*12;
  float m = mrow[0];
  int bi=(int)mrow[1], a=(int)mrow[2], gj=(int)mrow[3], gi=(int)mrow[4];
  float gtx=mrow[5], gty=mrow[6], gtw=mrow[7], gth=mrow[8];
  int cl=(int)mrow[9];
  const float* p = pv + (size_t)cand*96;
  float dx = 2.f*sigmf(p[0]) - 0.5f;
  float dy = 2.f*sigmf(p[1]) - 0.5f;
  float sw = 2.f*sigmf(p[2]), sh = 2.f*sigmf(p[3]);
  float dw = sw*sw*c_anchors[lvl][a][0];
  float dh = sh*sh*c_anchors[lvl][a][1];
  float iou = ciou_f(dx,dy,dw,dh, gtx,gty,gtw,gth);
  atomicAdd(ws + ACC_REG + lvl, (1.f - iou)*m);
  float iou_d = fmaxf(iou, 0.f)*m;      // >= 0 -> uint compare is monotone
  float* pcell = pos + ((size_t)(bi*3 + a)*Wd*Wd + (size_t)gj*Wd + gi);
  atomicMax((unsigned int*)pcell, __float_as_uint(iou_d));
  float s = 0.f;
  for (int ch=0; ch<80; ch++) s += softplusf(p[5+ch]);
  s -= p[5+cl];                          // - x*z with one-hot target (CP=1,CN=0)
  atomicAdd(ws + ACC_CLS + lvl, s*m);
}

// ---- sparse objectness correction: sum p_obj * pos ----
__global__ void obj_corr_kernel(const float* __restrict__ pobj, const float* __restrict__ pos,
                                float* __restrict__ acc, int N){
  int i = blockIdx.x*blockDim.x + threadIdx.x;
  float v = (i < N) ? pobj[i]*pos[i] : 0.f;
  for (int off=16; off>0; off>>=1) v += __shfl_xor(v, off, 32);
  if ((threadIdx.x & 31) == 0) atomicAdd(acc, v);
}

__global__ void final_kernel(const float* __restrict__ ws, float* __restrict__ out){
  if (threadIdx.x || blockIdx.x) return;
  const float BAL[3] = {4.f, 1.f, 0.4f};
  const int   HWs[3] = {6400, 1600, 400};
  float reg=0.f, cls=0.f, obj=0.f;
  for (int i=0;i<3;i++){
    float cnt = fmaxf(ws[ACC_CNT+i], 1.f);
    reg += ws[ACC_REG+i]/cnt;
    cls += ws[ACC_CLS+i]/(cnt*80.f);
    float cells = 16.f*3.f*(float)HWs[i];
    obj += (ws[ACC_SP+i] - ws[ACC_CORR+i])/cells * BAL[i];
  }
  out[0] = reg*0.05f + obj*1.0f + cls*0.5f;
}

extern "C" void kernel_launch(void* const* d_in, const int* in_sizes, int n_in,
                              void* d_out, int out_size, void* d_ws, size_t ws_size,
                              hipStream_t stream){
  (void)in_sizes; (void)n_in; (void)out_size; (void)ws_size;
  const float* f0 = (const float*)d_in[0];
  const float* f1 = (const float*)d_in[1];
  const float* f2 = (const float*)d_in[2];
  const float* w0 = (const float*)d_in[3];
  const float* b0 = (const float*)d_in[4];
  const float* w1 = (const float*)d_in[5];
  const float* b1 = (const float*)d_in[6];
  const float* w2 = (const float*)d_in[7];
  const float* b2 = (const float*)d_in[8];
  const float* gtb = (const float*)d_in[9];
  const int*   gtc = (const int*)d_in[10];
  const int*   img = (const int*)d_in[11];
  float* ws  = (float*)d_ws;
  float* out = (float*)d_out;

  init_kernel<<<(ZERO_END+255)/256, 256, 0, stream>>>(ws);

  // memory-bound streaming passes (dominant cost ~183MB -> ~8us at 23.3TB/s)
  obj_map_kernel<<<800, 256, 0, stream>>>(f0, w0, b0, ws+POBJ0, ws+ACC_SP+0, 256, 6400);
  obj_map_kernel<<<200, 256, 0, stream>>>(f1, w1, b1, ws+POBJ1, ws+ACC_SP+1, 512, 1600);
  obj_map_kernel<<< 50, 256, 0, stream>>>(f2, w2, b2, ws+POBJ2, ws+ACC_SP+2, 1024, 400);

  targets_kernel<<<90, 256, 0, stream>>>(gtb, gtc, img, ws);

  pos_gemm_kernel<<<360, 256, 0, stream>>>(f0, w0, b0, ws+META0, ws+PVOFF, 256, 6400, 80);
  loss_pos_kernel<<<30, 256, 0, stream>>>(ws+META0, ws+PVOFF, ws+POS0, ws, 0, 80);

  pos_gemm_kernel<<<360, 256, 0, stream>>>(f1, w1, b1, ws+META1, ws+PVOFF, 512, 1600, 40);
  loss_pos_kernel<<<30, 256, 0, stream>>>(ws+META1, ws+PVOFF, ws+POS1, ws, 1, 40);

  pos_gemm_kernel<<<360, 256, 0, stream>>>(f2, w2, b2, ws+META2, ws+PVOFF, 1024, 400, 20);
  loss_pos_kernel<<<30, 256, 0, stream>>>(ws+META2, ws+PVOFF, ws+POS2, ws, 2, 20);

  obj_corr_kernel<<<1200, 256, 0, stream>>>(ws+POBJ0, ws+POS0, ws+ACC_CORR+0, 307200);
  obj_corr_kernel<<< 300, 256, 0, stream>>>(ws+POBJ1, ws+POS1, ws+ACC_CORR+1,  76800);
  obj_corr_kernel<<<  75, 256, 0, stream>>>(ws+POBJ2, ws+POS2, ws+ACC_CORR+2,  19200);

  final_kernel<<<1, 1, 0, stream>>>(ws, out);
}
